// MovementPredictionHead_22067541966986
// MI455X (gfx1250) — matrix operand order repaired and checked by reading, not compiled, not run on hardware
//
#include <hip/hip_runtime.h>
#include <hip/hip_bf16.h>

typedef __attribute__((ext_vector_type(16))) __bf16 v16bf;
typedef __attribute__((ext_vector_type(8)))  __bf16 v8bf;
typedef __attribute__((ext_vector_type(8)))  float  v8f;
typedef __attribute__((ext_vector_type(4)))  float  f4v;

constexpr int Bc = 4, Nc = 512, Ec = 512, Hc = 32, Pc = 128, Dc = 16;
constexpr float SCALING = 0.25f;   // D^-0.5, D=16
constexpr float EPSc = 1e-5f;

__device__ __forceinline__ __bf16 f2bf(float f) {
    unsigned u = __builtin_bit_cast(unsigned, f);
    unsigned r = (u + 0x7FFFu + ((u >> 16) & 1u)) >> 16;
    return __builtin_bit_cast(__bf16, (unsigned short)r);
}

__device__ __forceinline__ v8f wmma_bf16(v16bf a, v16bf b, v8f c) {
    return __builtin_amdgcn_wmma_f32_16x16x32_bf16(false, a, false, b, (short)0, c,
                                                   false, false);
}

// ---------------- Kernel 1: layernorm(query) -> bf16 ----------------
__global__ void ln_query_kernel(const float* __restrict__ q,
                                const float* __restrict__ g,
                                const float* __restrict__ bta,
                                __bf16* __restrict__ xbf) {
    int row = blockIdx.x;            // 0 .. B*N-1
    int t = threadIdx.x;             // 0 .. 127
    const float* rp = q + (size_t)row * Ec;
    float4 vv = ((const float4*)rp)[t];
    float s  = vv.x + vv.y + vv.z + vv.w;
    float sq = vv.x*vv.x + vv.y*vv.y + vv.z*vv.z + vv.w*vv.w;
    __shared__ float rs[128], rq[128];
    rs[t] = s; rq[t] = sq; __syncthreads();
    for (int off = 64; off > 0; off >>= 1) {
        if (t < off) { rs[t] += rs[t+off]; rq[t] += rq[t+off]; }
        __syncthreads();
    }
    float mu  = rs[0] * (1.0f / Ec);
    float var = rq[0] * (1.0f / Ec) - mu * mu;
    float rsg = rsqrtf(var + EPSc);
    int c0 = t * 4;
    __bf16* op = xbf + (size_t)row * Ec + c0;
    op[0] = f2bf((vv.x - mu) * rsg * g[c0+0] + bta[c0+0]);
    op[1] = f2bf((vv.y - mu) * rsg * g[c0+1] + bta[c0+1]);
    op[2] = f2bf((vv.z - mu) * rsg * g[c0+2] + bta[c0+2]);
    op[3] = f2bf((vv.w - mu) * rsg * g[c0+3] + bta[c0+3]);
}

// ---------------- Kernel 2: convert weights to bf16 ----------------
__global__ void cvt_w_kernel(const float* __restrict__ Wq, const float* __restrict__ Wk,
                             const float* __restrict__ Wv, const float* __restrict__ Wb,
                             __bf16* __restrict__ out) {
    const int EE = Ec * Ec;
    int idx = blockIdx.x * 256 + threadIdx.x;
    int total = 3 * EE + Hc * Pc;
    if (idx >= total) return;
    float v;
    if      (idx <     EE) v = Wq[idx];
    else if (idx < 2 * EE) v = Wk[idx - EE];
    else if (idx < 3 * EE) v = Wv[idx - 2 * EE];
    else                   v = Wb[idx - 3 * EE];
    out[idx] = f2bf(v);
}

// ---------------- Kernel 3: QKV projection GEMM (WMMA) ----------------
// out = x @ W^T ; one wave computes a 16x16 tile, K = 512 in 16 WMMA steps.
__global__ void qkv_kernel(const __bf16* __restrict__ xbf, const __bf16* __restrict__ wbuf,
                           __bf16* __restrict__ qb, __bf16* __restrict__ kb,
                           __bf16* __restrict__ vT) {
    int w = blockIdx.x;
    int which = w % 3; int rest = w / 3;
    int ot = rest % (Ec / 16);
    int mt = rest / (Ec / 16);
    int l = threadIdx.x, hl = l >> 4, ll = l & 15;
    const __bf16* W = wbuf + (size_t)which * Ec * Ec;
    v8f acc; for (int i = 0; i < 8; i++) acc[i] = 0.f;
    const __bf16* arow = xbf + (size_t)(mt * 16 + ll) * Ec;
    const __bf16* brow = W   + (size_t)(ot * 16 + ll) * Ec;
    for (int k0 = 0; k0 < Ec; k0 += 32) {
        v8bf a0 = *(const v8bf*)(arow + k0 + hl * 8);
        v8bf a1 = *(const v8bf*)(arow + k0 + 16 + hl * 8);
        v8bf b0 = *(const v8bf*)(brow + k0 + hl * 8);
        v8bf b1 = *(const v8bf*)(brow + k0 + 16 + hl * 8);
        v16bf af, bf_;
        for (int i = 0; i < 8; i++) { af[i] = a0[i]; af[8+i] = a1[i];
                                      bf_[i] = b0[i]; bf_[8+i] = b1[i]; }
        acc = wmma_bf16(af, bf_, acc);
    }
    int h = ot, d = ll;
    for (int r = 0; r < 8; r++) {
        int m = mt * 16 + r + 8 * hl;
        int b = m / Nc, n = m % Nc;
        float v = acc[r];
        if      (which == 0) qb[((size_t)(b * Hc + h) * Nc + n) * Dc + d] = f2bf(v * SCALING);
        else if (which == 1) kb[((size_t)(b * Hc + h) * Nc + n) * Dc + d] = f2bf(v);
        else                 vT[((size_t)(b * Hc + h) * Dc + d) * Nc + n] = f2bf(v);
    }
}

// ---------------- Kernel 4: fused pair-LN + bias GEMM (WMMA) ----------------
// block = 64 threads (2 waves); handles 16 j-rows for one (b, i); output 16j x 32h.
// pair/mask are streamed once -> non-temporal loads keep L2 free for the bias
// intermediate, which the attention kernel consumes next.
__global__ void bias_kernel(const float* __restrict__ pair, const float* __restrict__ pg,
                            const float* __restrict__ pb, const __bf16* __restrict__ wb_bf,
                            const float* __restrict__ bbv, const float* __restrict__ mask,
                            float* __restrict__ bias) {
    int blk = blockIdx.x;                  // ((b*N + i) * N/16) + jt
    int jt = blk % (Nc / 16); int bi = blk / (Nc / 16);
    int i = bi % Nc, b = bi / Nc;
    int t = threadIdx.x;
    int jr = t >> 2, qd = t & 3;           // row in tile, quarter of 128 cols
    __shared__ float redS[64], redQ[64];
    __shared__ float muA[16], rsA[16];
    __shared__ __align__(32) __bf16 sA[16 * 128];

    const float* rowp = pair + (((size_t)(b * Nc + i)) * Nc + (jt * 16 + jr)) * Pc;
    const f4v* rp4 = (const f4v*)rowp + qd * 8;
    f4v vv[8];
    float s = 0.f, sq = 0.f;
    for (int u = 0; u < 8; u++) {
        vv[u] = __builtin_nontemporal_load(rp4 + u);     // streamed once: NT hint
        s  += vv[u][0] + vv[u][1] + vv[u][2] + vv[u][3];
        sq += vv[u][0]*vv[u][0] + vv[u][1]*vv[u][1] + vv[u][2]*vv[u][2] + vv[u][3]*vv[u][3];
    }
    redS[t] = s; redQ[t] = sq; __syncthreads();
    if (qd == 0) {
        float S = redS[t] + redS[t+1] + redS[t+2] + redS[t+3];
        float Q = redQ[t] + redQ[t+1] + redQ[t+2] + redQ[t+3];
        float mu = S * (1.f / Pc);
        float var = Q * (1.f / Pc) - mu * mu;
        muA[jr] = mu; rsA[jr] = rsqrtf(var + EPSc);
    }
    __syncthreads();
    float mu = muA[jr], rsg = rsA[jr];
    int c0 = qd * 32;
    for (int u = 0; u < 8; u++) {
        int c = c0 + u * 4;
        sA[jr * 128 + c + 0] = f2bf((vv[u][0] - mu) * rsg * pg[c+0] + pb[c+0]);
        sA[jr * 128 + c + 1] = f2bf((vv[u][1] - mu) * rsg * pg[c+1] + pb[c+1]);
        sA[jr * 128 + c + 2] = f2bf((vv[u][2] - mu) * rsg * pg[c+2] + pb[c+2]);
        sA[jr * 128 + c + 3] = f2bf((vv[u][3] - mu) * rsg * pg[c+3] + pb[c+3]);
    }
    __syncthreads();
    // GEMM: (16 x 128) @ (128 x 16 heads) per wave; 2 waves -> 32 heads
    int wv = t >> 5, l = t & 31, hl = l >> 4, ll = l & 15;
    v8f acc; for (int u = 0; u < 8; u++) acc[u] = 0.f;
    const __bf16* brow = wb_bf + (size_t)(wv * 16 + ll) * Pc;
    for (int ks = 0; ks < 4; ks++) {
        int k0 = ks * 32;
        v8bf a0 = *(const v8bf*)(sA + ll * 128 + k0 + hl * 8);
        v8bf a1 = *(const v8bf*)(sA + ll * 128 + k0 + 16 + hl * 8);
        v8bf b0 = *(const v8bf*)(brow + k0 + hl * 8);
        v8bf b1 = *(const v8bf*)(brow + k0 + 16 + hl * 8);
        v16bf af, bf_;
        for (int u = 0; u < 8; u++) { af[u] = a0[u]; af[8+u] = a1[u];
                                      bf_[u] = b0[u]; bf_[8+u] = b1[u]; }
        acc = wmma_bf16(af, bf_, acc);
    }
    for (int r = 0; r < 8; r++) {
        int h = wv * 16 + ll;
        int j = jt * 16 + r + 8 * hl;
        size_t idx = (((size_t)(b * Hc + h)) * Nc + i) * Nc + j;
        float mk = __builtin_nontemporal_load(mask + idx);  // read once: NT hint
        bias[idx] = acc[r] + bbv[h] + mk;                   // RT store -> L2 resident
    }
}

// ---------------- Kernel 5: flash attention + rotary-delta contraction ----------------
// one wave per (b, h, 16-row i tile). K-dim = D = 16, zero-padded to 32 for WMMA.
__global__ void attn_kernel(const __bf16* __restrict__ qb, const __bf16* __restrict__ kb,
                            const __bf16* __restrict__ vT, const float* __restrict__ bias,
                            const float* __restrict__ delta, float* __restrict__ xo) {
    int it = blockIdx.x % (Nc / 16);
    int bh = blockIdx.x / (Nc / 16);
    int b = bh / Hc, h = bh % Hc;
    int l = threadIdx.x, hl = l >> 4, ll = l & 15;
    __shared__ __align__(16) float sP[256];
    const __bf16 zbf = __builtin_bit_cast(__bf16, (unsigned short)0);

    v16bf qf;
    {
        const __bf16* qp = qb + ((size_t)bh * Nc + it * 16 + ll) * Dc + hl * 8;
        v8bf q8 = *(const v8bf*)qp;
        for (int u = 0; u < 8; u++) { qf[u] = q8[u]; qf[8+u] = zbf; }  // K 16..31 = 0
    }
    float mrow[8], lrow[8];
    v8f acc0, acc1, acc2;
    for (int r = 0; r < 8; r++) { mrow[r] = -1e30f; lrow[r] = 0.f;
                                  acc0[r] = 0.f; acc1[r] = 0.f; acc2[r] = 0.f; }

    for (int jt = 0; jt < Nc / 16; jt++) {
        // K fragment: lane (col) = j token, d = hl*8 + u in slots 0..7; slots 8..15 zero
        v16bf kf; for (int u = 0; u < 16; u++) kf[u] = zbf;
        {
            const __bf16* kp = kb + ((size_t)bh * Nc + jt * 16 + ll) * Dc;
            v8bf k0 = *(const v8bf*)(kp + hl * 8);
            for (int u = 0; u < 8; u++) { kf[u] = k0[u]; }
        }
        v8f sv; for (int r = 0; r < 8; r++) sv[r] = 0.f;
        sv = wmma_bf16(qf, kf, sv);

        const float* bp = bias + ((size_t)bh * Nc + it * 16) * Nc + jt * 16 + ll;
        for (int r = 0; r < 8; r++) {
            int row = r + 8 * hl;
            float sl = sv[r] + bp[(size_t)row * Nc];
            float mx = sl;
            for (int off = 1; off < 16; off <<= 1) mx = fmaxf(mx, __shfl_xor(mx, off, 16));
            float mn = fmaxf(mrow[r], mx);
            float p  = __expf(sl - mn);
            float sc = __expf(mrow[r] - mn);
            float ps = p;
            for (int off = 1; off < 16; off <<= 1) ps += __shfl_xor(ps, off, 16);
            lrow[r] = lrow[r] * sc + ps;
            mrow[r] = mn;
            acc0[r] *= sc; acc1[r] *= sc; acc2[r] *= sc;
            sP[row * 16 + ll] = p;
        }
        __syncthreads();

        // delta_pos for this lane's A-fragment slots (row = ll, j = hl*8 + e)
        float dl[8][3];
        {
            int irow = it * 16 + ll;
            const float* dp = delta + (((size_t)(b * Nc + irow)) * Nc + jt * 16 + hl * 8) * 3;
            for (int e = 0; e < 8; e++) {
                dl[e][0] = dp[e*3+0]; dl[e][1] = dp[e*3+1]; dl[e][2] = dp[e*3+2];
            }
        }
        // V fragment from transposed v: lane (col) = d, j = hl*8 + u in slots 0..7; slots 8..15 zero
        v16bf vf; for (int u = 0; u < 16; u++) vf[u] = zbf;
        {
            const __bf16* vp = vT + ((size_t)bh * Dc + ll) * Nc + jt * 16;
            v8bf v0 = *(const v8bf*)(vp + hl * 8);
            for (int u = 0; u < 8; u++) { vf[u] = v0[u]; }
        }
        for (int c = 0; c < 3; c++) {
            v16bf pf;
            for (int e = 0; e < 8; e++) {
                float pv = sP[ll * 16 + hl * 8 + e];
                pf[e] = f2bf(pv * dl[e][c]);
                pf[8+e] = zbf;                       // K 16..31 = 0
            }
            if      (c == 0) acc0 = wmma_bf16(pf, vf, acc0);
            else if (c == 1) acc1 = wmma_bf16(pf, vf, acc1);
            else             acc2 = wmma_bf16(pf, vf, acc2);
        }
        __syncthreads();
    }
    for (int r = 0; r < 8; r++) {
        int n = it * 16 + r + 8 * hl;
        float inv = 1.0f / lrow[r];
        int e = h * 16 + ll;
        size_t base = ((size_t)(b * Nc + n)) * 3;
        xo[(base + 0) * Ec + e] = acc0[r] * inv;
        xo[(base + 1) * Ec + e] = acc1[r] * inv;
        xo[(base + 2) * Ec + e] = acc2[r] * inv;
    }
}

// ---------------- Kernel 6: final force projection ----------------
__global__ void force_kernel(const float* __restrict__ xo, const float* __restrict__ Wf,
                             float* __restrict__ out) {
    int o = blockIdx.x;        // 0 .. B*N*3-1
    int t = threadIdx.x;       // 0 .. 63
    const float* rp = xo + (size_t)o * Ec;
    float s = 0.f;
    for (int u = 0; u < 8; u++) { int c = t * 8 + u; s += rp[c] * Wf[c]; }
    __shared__ float red[64];
    red[t] = s; __syncthreads();
    for (int off = 32; off > 0; off >>= 1) {
        if (t < off) red[t] += red[t + off];
        __syncthreads();
    }
    if (t == 0) out[o] = red[0];
}

extern "C" void kernel_launch(void* const* d_in, const int* in_sizes, int n_in,
                              void* d_out, int out_size, void* d_ws, size_t ws_size,
                              hipStream_t stream) {
    const float* query = (const float*)d_in[0];
    const float* pair  = (const float*)d_in[1];
    const float* mask  = (const float*)d_in[2];
    const float* delta = (const float*)d_in[3];
    const float* ln_g  = (const float*)d_in[4];
    const float* ln_b  = (const float*)d_in[5];
    const float* Wq    = (const float*)d_in[6];
    const float* Wk    = (const float*)d_in[7];
    const float* Wv    = (const float*)d_in[8];
    const float* pg    = (const float*)d_in[9];
    const float* pb    = (const float*)d_in[10];
    const float* Wb    = (const float*)d_in[11];
    const float* bbv   = (const float*)d_in[12];
    const float* Wf    = (const float*)d_in[13];

    char* ws = (char*)d_ws;
    size_t off = 0;
    auto carve = [&](size_t bytes) -> void* {
        void* p = ws + off;
        off += (bytes + 255) & ~(size_t)255;
        return p;
    };
    __bf16* xbf  = (__bf16*)carve((size_t)Bc * Nc * Ec * 2);
    __bf16* wbuf = (__bf16*)carve(((size_t)3 * Ec * Ec + Hc * Pc) * 2);
    __bf16* qb   = (__bf16*)carve((size_t)Bc * Hc * Nc * Dc * 2);
    __bf16* kb   = (__bf16*)carve((size_t)Bc * Hc * Nc * Dc * 2);
    __bf16* vT   = (__bf16*)carve((size_t)Bc * Hc * Nc * Dc * 2);
    float*  bias = (float*)carve((size_t)Bc * Hc * Nc * Nc * 4);
    float*  xo   = (float*)carve((size_t)Bc * Nc * 3 * Ec * 4);

    ln_query_kernel<<<Bc * Nc, 128, 0, stream>>>(query, ln_g, ln_b, xbf);

    int total = 3 * Ec * Ec + Hc * Pc;
    cvt_w_kernel<<<(total + 255) / 256, 256, 0, stream>>>(Wq, Wk, Wv, Wb, wbuf);

    qkv_kernel<<<(Bc * Nc / 16) * (Ec / 16) * 3, 32, 0, stream>>>(xbf, wbuf, qb, kb, vT);

    bias_kernel<<<Bc * Nc * (Nc / 16), 64, 0, stream>>>(pair, pg, pb,
                                                        wbuf + (size_t)3 * Ec * Ec,
                                                        bbv, mask, bias);

    attn_kernel<<<Bc * Hc * (Nc / 16), 32, 0, stream>>>(qb, kb, vT, bias, delta, xo);

    force_kernel<<<Bc * Nc * 3, 64, 0, stream>>>(xo, Wf, (float*)d_out);
}
